// DPLSTM_10737418240033
// MI455X (gfx1250) — compile-verified, matrix operand order checked by reading, not run on hardware
//
#include <hip/hip_runtime.h>

// ---------------------------------------------------------------------------
// LSTM  (T=2048, B=64, I=H=256) for gfx1250 / MI455X.
//   K0: prep    - w_ih/w_hh f32->bf16, bias = b_ih+b_hh, zero h0 / sync ctr
//   K1: x_gates - bf16 WMMA GEMM [131072,256]x[256,1024] -> f32 in ws
//                 (B fragments register-resident across 8 M-tiles per block)
//   K2: scan    - 16 persistent blocks, per-step h@w_hh^T via bf16 WMMA,
//                 w_hh slice staged to LDS via global_load_async_to_lds,
//                 c-state resident in VGPRs, h handoff via L2 + atomic ctr.
// ---------------------------------------------------------------------------

#define T_    2048
#define B_    64
#define I_    256
#define H_    256
#define FOURH 1024

typedef __attribute__((ext_vector_type(16))) __bf16 v16bf;
typedef __attribute__((ext_vector_type(8)))  __bf16 v8bf;
typedef __attribute__((ext_vector_type(8)))  float  v8f;

__device__ __forceinline__ unsigned short f32_to_bf16_bits(float f) {
  unsigned int u = __builtin_bit_cast(unsigned int, f);
  unsigned int r = u + 0x7fffu + ((u >> 16) & 1u);       // round-to-nearest-even
  if ((u & 0x7f800000u) == 0x7f800000u) r = u;           // keep inf/nan
  return (unsigned short)(r >> 16);
}

__device__ __forceinline__ float fast_sigmoid(float x) {
  return 1.0f / (1.0f + __expf(-x));
}
__device__ __forceinline__ float fast_tanh(float x) {
  // 2*sigmoid(2x)-1 : saturates cleanly to +-1, no inf/inf NaN path
  return 2.0f / (1.0f + __expf(-2.0f * x)) - 1.0f;
}

// Build a 16x32 bf16 A/B fragment for v_wmma_f32_16x16x32_bf16.
// Per ISA 7.12.2: lanes<16 hold K {ks..ks+7, ks+16..ks+23} with ks=k0*32,
// lanes>=16 the same with ks=k0*32+8.  base = start of this lane's row.
__device__ __forceinline__ v16bf make_frag(const unsigned short* base, int ks) {
  v8bf lo = *reinterpret_cast<const v8bf*>(base + ks);
  v8bf hi = *reinterpret_cast<const v8bf*>(base + ks + 16);
  v16bf r;
#pragma unroll
  for (int i = 0; i < 8; ++i) { r[i] = lo[i]; r[8 + i] = hi[i]; }
  return r;
}

// --------------------------- K0: prep --------------------------------------
__global__ void lstm_prep_kernel(const float* __restrict__ wih,
                                 const float* __restrict__ whh,
                                 const float* __restrict__ bih,
                                 const float* __restrict__ bhh,
                                 unsigned short* __restrict__ wih_bf,
                                 unsigned short* __restrict__ whh_bf,
                                 float* __restrict__ bias,
                                 unsigned short* __restrict__ hbuf,
                                 unsigned int* __restrict__ sync) {
  const int tid    = blockIdx.x * blockDim.x + threadIdx.x;
  const int stride = gridDim.x * blockDim.x;
  for (int i = tid; i < FOURH * I_; i += stride) {
    wih_bf[i] = f32_to_bf16_bits(wih[i]);
    whh_bf[i] = f32_to_bf16_bits(whh[i]);
  }
  if (tid < FOURH) bias[tid] = bih[tid] + bhh[tid];
  for (int i = tid; i < B_ * H_; i += stride) hbuf[i] = 0;  // h0 = 0 (buffer 0)
  if (tid == 0) *sync = 0u;                                  // reset per launch
}

// --------------------------- K1: x_gates GEMM ------------------------------
// grid (M/128, N/128), block 256 (8 waves). Wave w owns N-tile blockIdx.y*8+w;
// its 8 K-chunk B fragments stay in registers while the block sweeps 8 M-tiles
// (x tile restaged to LDS as bf16 each iteration).
__global__ void __launch_bounds__(256) lstm_xgates_kernel(
    const float* __restrict__ x,
    const unsigned short* __restrict__ wih_bf,
    const float* __restrict__ bias,
    float* __restrict__ xg) {
  __shared__ __align__(16) unsigned short sA[16 * 256];   // 8 KB bf16 A tile

  const int lane  = threadIdx.x & 31;
  const int wave  = threadIdx.x >> 5;
  const int ntile = blockIdx.y * 8 + wave;
  const int ncol  = ntile * 16 + (lane & 15);
  const int sel   = (lane < 16) ? 0 : 8;

  // B fragments (w_ih row = output column), register resident: 64 VGPRs
  const unsigned short* brow = wih_bf + (size_t)ncol * I_;
  v16bf bfrag[8];
#pragma unroll
  for (int k0 = 0; k0 < 8; ++k0) bfrag[k0] = make_frag(brow, k0 * 32 + sel);

  const float bv     = bias[ncol];
  const int   rowadd = (lane < 16) ? 0 : 8;
  const int   rstg   = threadIdx.x >> 4;          // staging row 0..15
  const int   cstg   = (threadIdx.x & 15) * 16;   // staging col 0..240
  const unsigned short* arow = sA + (lane & 15) * 256;

  for (int mt = 0; mt < 8; ++mt) {
    const int mtile = blockIdx.x * 8 + mt;

    // stage A: rows m = mtile*16 + 0..15, 256 f32 -> bf16 (packed dword store)
    {
      const float* src = x + ((size_t)mtile * 16 + rstg) * I_ + cstg;
      unsigned int* dst = reinterpret_cast<unsigned int*>(sA + rstg * 256 + cstg);
#pragma unroll
      for (int i = 0; i < 16; i += 4) {
        float4 v = *reinterpret_cast<const float4*>(src + i);
        dst[i / 2 + 0] = (unsigned int)f32_to_bf16_bits(v.x) |
                         ((unsigned int)f32_to_bf16_bits(v.y) << 16);
        dst[i / 2 + 1] = (unsigned int)f32_to_bf16_bits(v.z) |
                         ((unsigned int)f32_to_bf16_bits(v.w) << 16);
      }
    }
    __syncthreads();

    v8f acc = {};
#pragma unroll
    for (int k0 = 0; k0 < 8; ++k0) {
      v16bf a = make_frag(arow, k0 * 32 + sel);
      acc = __builtin_amdgcn_wmma_f32_16x16x32_bf16(false, a, false, bfrag[k0],
                                                    (short)0, acc, false, false);
    }

    const size_t mbase = (size_t)mtile * 16;
#pragma unroll
    for (int r = 0; r < 8; ++r) {
      xg[(mbase + r + rowadd) * FOURH + ncol] = acc[r] + bv;
    }
    __syncthreads();   // A tile consumed before restage
  }
}

// --------------------------- K2: recurrent scan ----------------------------
// 16 blocks x 128 threads (4 waves). Block g owns h-cols [16g,16g+16);
// wave w owns batch rows [16w,16w+16). c-state lives in VGPRs for all T steps.
__global__ void __launch_bounds__(128, 1) lstm_scan_kernel(
    const unsigned short* __restrict__ whh_bf,
    const float* __restrict__ xg,
    unsigned short* __restrict__ hbuf,     // 2 x [64][256] bf16, double buffer
    unsigned int* __restrict__ sync,
    float* __restrict__ out) {
  __shared__ __align__(16) unsigned short sB[64 * 256];   // 32 KB w_hh slice

  const int g     = blockIdx.x;          // 0..15
  const int jbase = g * 16;
  const int lane  = threadIdx.x & 31;
  const int wave  = threadIdx.x >> 5;    // M-tile (batch rows 16w..16w+15)

  // Stage the 64 w_hh rows this block needs (gate q, col j -> row q*256+jb+j)
  // via CDNA5 async global->LDS copy.  A generic pointer into the LDS aperture
  // truncates to the LDS byte offset (ISA 10.2), usable as the VDST address.
  {
    const int r    = threadIdx.x >> 1;   // 0..63 local row
    const int half = threadIdx.x & 1;
    const int q    = r >> 4;
    const int j    = jbase + (r & 15);
    const unsigned short* src = whh_bf + ((size_t)(q * H_ + j)) * H_ + half * 128;
    unsigned short*       dst = sB + r * 256 + half * 128;
    unsigned int       ldsa = (unsigned int)(size_t)dst;
    unsigned long long ga   = (unsigned long long)(size_t)src;
#pragma unroll
    for (int i = 0; i < 128; i += 8) {   // 16 x 16B per thread = 32 KB total
      unsigned int       l  = ldsa + (unsigned int)(i * 2);
      unsigned long long gv = ga + (unsigned long long)(i * 2);
      asm volatile("global_load_async_to_lds_b128 %0, %1, off"
                   :: "v"(l), "v"(gv) : "memory");
    }
    asm volatile("s_wait_asynccnt 0" ::: "memory");
  }
  __syncthreads();

  const int nloc   = lane & 15;                 // local output col
  const int sel    = (lane < 16) ? 0 : 8;
  const int rowadd = (lane < 16) ? 0 : 8;
  const int mrow   = wave * 16 + nloc;          // this lane's A (h) row
  const int col    = jbase + nloc;              // global h column

  v8f c  = {};
  v8f hv = {};

  for (int t = 0; t < T_; ++t) {
    if (t > 0) {  // grid barrier: everyone's h(t-1) slice visible in L2
      __syncthreads();
      if (threadIdx.x == 0) {
        __threadfence();
        atomicAdd(sync, 1u);
        const unsigned int target = (unsigned int)(16 * t);
        while (__hip_atomic_load(sync, __ATOMIC_RELAXED,
                                 __HIP_MEMORY_SCOPE_AGENT) < target) {
          __builtin_amdgcn_s_sleep(1);
        }
        __threadfence();
      }
      __syncthreads();
    }

    const unsigned short* hprev = hbuf + (size_t)(t & 1) * (B_ * H_);
    const float*          xgt   = xg + (size_t)t * (B_ * FOURH);

    // init accumulators with precomputed x_gates (+bias)
    v8f acc[4];
#pragma unroll
    for (int q = 0; q < 4; ++q) {
      const int ncol = q * H_ + col;
#pragma unroll
      for (int r = 0; r < 8; ++r)
        acc[q][r] = xgt[(size_t)(wave * 16 + r + rowadd) * FOURH + ncol];
    }

    const unsigned short* arow = hprev + (size_t)mrow * H_;
#pragma unroll
    for (int k0 = 0; k0 < 8; ++k0) {
      v16bf a = make_frag(arow, k0 * 32 + sel);
#pragma unroll
      for (int q = 0; q < 4; ++q) {
        v16bf b = make_frag(sB + (q * 16 + nloc) * 256, k0 * 32 + sel);
        acc[q] = __builtin_amdgcn_wmma_f32_16x16x32_bf16(
            false, a, false, b, (short)0, acc[q], false, false);
      }
    }

    // gates -> activations -> state update (all in registers), publish h
    unsigned short* hnext = hbuf + (size_t)((t + 1) & 1) * (B_ * H_);
    float*          hs_t  = out + (size_t)t * (B_ * H_);
#pragma unroll
    for (int r = 0; r < 8; ++r) {
      const float it = fast_sigmoid(acc[0][r]);
      const float ft = fast_sigmoid(acc[1][r]);
      const float gt = fast_tanh(acc[2][r]);
      const float ot = fast_sigmoid(acc[3][r]);
      const float cr = ft * c[r] + it * gt;
      c[r] = cr;
      const float hr = ot * fast_tanh(cr);
      hv[r] = hr;
      const int brow = wave * 16 + r + rowadd;
      hnext[brow * H_ + col] = f32_to_bf16_bits(hr);
      hs_t[brow * H_ + col]  = hr;
    }
  }

  // final states: out layout = hs | h_T | c_T
  float* hT = out + (size_t)T_ * B_ * H_;
  float* cT = hT + B_ * H_;
#pragma unroll
  for (int r = 0; r < 8; ++r) {
    const int brow = wave * 16 + r + rowadd;
    hT[brow * H_ + col] = hv[r];
    cT[brow * H_ + col] = c[r];
  }
}

// --------------------------- launcher --------------------------------------
extern "C" void kernel_launch(void* const* d_in, const int* in_sizes, int n_in,
                              void* d_out, int out_size, void* d_ws,
                              size_t ws_size, hipStream_t stream) {
  const float* x   = (const float*)d_in[0];
  const float* wih = (const float*)d_in[1];
  const float* whh = (const float*)d_in[2];
  const float* bih = (const float*)d_in[3];
  const float* bhh = (const float*)d_in[4];
  float* out = (float*)d_out;

  char* ws = (char*)d_ws;
  // ws layout (requires ~515 MB):
  unsigned short* wih_bf = (unsigned short*)(ws);                    // 512 KB
  unsigned short* whh_bf = (unsigned short*)(ws + (512 << 10));      // 512 KB
  float*          bias   = (float*)(ws + (1 << 20));                 // 4 KB
  unsigned int*   syncc  = (unsigned int*)(ws + (1 << 20) + 4096);   // 4 B
  unsigned short* hbuf   = (unsigned short*)(ws + (1 << 20) + 8192); // 64 KB
  float*          xg     = (float*)(ws + (2 << 20));                 // 512 MB

  hipLaunchKernelGGL(lstm_prep_kernel, dim3(256), dim3(256), 0, stream,
                     wih, whh, bih, bhh, wih_bf, whh_bf, bias, hbuf, syncc);

  hipLaunchKernelGGL(lstm_xgates_kernel, dim3((T_ * B_) / 128, FOURH / 128),
                     dim3(256), 0, stream, x, wih_bf, bias, xg);

  hipLaunchKernelGGL(lstm_scan_kernel, dim3(16), dim3(128), 0, stream,
                     whh_bf, xg, hbuf, syncc, out);
}
